// CapsuleNet_10316511445838
// MI455X (gfx1250) — matrix operand order repaired and checked by reading, not compiled, not run on hardware
//
#include <hip/hip_runtime.h>
#include <hip/hip_bf16.h>

typedef __attribute__((ext_vector_type(16))) _Float16 v16h;
typedef __attribute__((ext_vector_type(8)))  _Float16 v8h;
typedef __attribute__((ext_vector_type(8)))  float    v8f;

#define N_IMG   512
#define NCAPS   1152     // 32*36 primary capsules
#define CAPD    8
#define OCAPS   10
#define K2      20736    // 81*256 reduction for conv2
#define COLS2   (N_IMG*36)   // 18432 GEMM columns for conv2

// LDS staging geometry for conv2 B tiles: 16 columns x 256 channels (f16),
// column pitch padded 256 -> 264 halves to spread LDS banks.
#define BCOL_PITCH 264
#define BUFH       (16 * BCOL_PITCH)     // halves per buffer (4224)

#define WAIT_ASYNC(n) asm volatile("s_wait_asynccnt " #n ::: "memory")

// ---------------------------------------------------------------------------
// Kernel 1: conv1 (1->256, 9x9, valid) + bias + relu, write f16 channels-last
// h1[((n*20+y)*20+x)*256 + c]
// ---------------------------------------------------------------------------
__global__ __launch_bounds__(256)
void k_conv1(const float* __restrict__ x, const float* __restrict__ w,
             const float* __restrict__ b, _Float16* __restrict__ h1) {
    __shared__ float sx[28 * 28];
    const int n = blockIdx.x;
    const int t = threadIdx.x;
    for (int e = t; e < 784; e += 256) sx[e] = x[n * 784 + e];
    __syncthreads();

    const int c = t;                       // one output channel per thread
    float wreg[81];
#pragma unroll
    for (int q = 0; q < 81; ++q) wreg[q] = w[c * 81 + q];
    const float bias = b[c];

    for (int y = 0; y < 20; ++y) {
        for (int xx = 0; xx < 20; ++xx) {
            float acc = bias;
#pragma unroll
            for (int kh = 0; kh < 9; ++kh)
#pragma unroll
                for (int kw = 0; kw < 9; ++kw)
                    acc += wreg[kh * 9 + kw] * sx[(y + kh) * 28 + (xx + kw)];
            acc = fmaxf(acc, 0.0f);
            h1[((n * 20 + y) * 20 + xx) * 256 + c] = (_Float16)acc;
        }
    }
}

// ---------------------------------------------------------------------------
// Kernel 2: repack conv2 weights [co][ci][kh][kw] f32 -> Wr[co][p*256+ci] f16,
// p = kh*9+kw  (A-matrix rows contiguous in reordered K)
// ---------------------------------------------------------------------------
__global__ __launch_bounds__(256)
void k_repack_w2(const float* __restrict__ w, _Float16* __restrict__ wr) {
    const int e = blockIdx.x * 256 + threadIdx.x;      // e < 256*20736
    const int co = e / K2;
    const int r  = e - co * K2;
    const int p  = r >> 8;                             // r / 256
    const int ci = r & 255;
    wr[e] = (_Float16)w[(co * 256 + ci) * 81 + p];
}

// ---------------------------------------------------------------------------
// Kernel 3: repack caps_w [i][k][o] f32 -> cw_t[i][o][k] f16 (k contiguous)
// ---------------------------------------------------------------------------
__global__ __launch_bounds__(256)
void k_repack_cw(const float* __restrict__ cw, _Float16* __restrict__ ct) {
    const int e = blockIdx.x * 256 + threadIdx.x;      // e < 1152*160*8
    const int i = e / (160 * 8);
    const int r = e - i * (160 * 8);
    const int o = r >> 3;
    const int k = r & 7;
    ct[e] = (_Float16)cw[i * (8 * 160) + k * 160 + o];
}

// ---------------------------------------------------------------------------
// Kernel 4: conv2 as implicit GEMM with v_wmma_f32_16x16x32_f16.
// One block = one 16-column tile (nt) x ALL 16 co-tiles; wave w accumulates
// co-tiles {w, w+8} (2 WMMA per K-step). Per (kh,kw) the shared B slab
// (16 cols x 256 ch f16 = 8KB) is staged to LDS with CDNA5 async loads,
// double-buffered: prefetch p+1 while computing p; s_wait_asynccnt 2 relies
// on in-order async-load completion. D tile rows = 2 capsules x 8 dims ->
// bias-added result stored directly as u_pre[n][i][0..7] (one v8f per lane).
// ---------------------------------------------------------------------------
__global__ __launch_bounds__(256)
void k_conv2_wmma(const _Float16* __restrict__ Wr, const _Float16* __restrict__ h1,
                  const float* __restrict__ bias, float* __restrict__ u_pre) {
    __shared__ _Float16 bstage[2 * BUFH];              // 2 x 8.25KB

    const int t    = threadIdx.x;
    const int lane = t & 31;
    const int wave = t >> 5;
    const int nt   = blockIdx.x;                       // 0..1151 column tiles
    const int lm   = lane & 15;
    const int hi   = lane >> 4;

    // ---- this lane's GEMM column (for WMMA + epilogue) ----
    const int j   = (nt << 4) + lm;
    const int n   = j / 36;
    const int pos = j - n * 36;
    const int oh  = pos / 6;
    const int ow  = pos - oh * 6;

    // ---- per-thread async-copy descriptors: chunks t and t+256 of 512 ----
    // chunk e: col = e>>5 (0..15), cg = e&31 (16B group of channels)
    int  ccol[2], ccg[2], cn[2], coh[2], cow[2];
#pragma unroll
    for (int q = 0; q < 2; ++q) {
        const int e  = t + q * 256;
        ccol[q] = e >> 5;
        ccg[q]  = e & 31;
        const int jc  = (nt << 4) + ccol[q];
        cn[q]  = jc / 36;
        const int pc  = jc - cn[q] * 36;
        coh[q] = pc / 6;
        cow[q] = pc - coh[q] * 6;
    }
    const unsigned lds_base =
        (unsigned)(unsigned long long)(__attribute__((address_space(3))) _Float16*)&bstage[0];

    // issue async copy of B slab for filter tap p into buffer `buf`
    auto issue_copy = [&](int p, int buf) {
        const int kh = p / 9;
        const int kw = p - kh * 9;
#pragma unroll
        for (int q = 0; q < 2; ++q) {
            const int y  = 2 * coh[q] + kh;
            const int xx = 2 * cow[q] + kw;
            const _Float16* gp =
                h1 + ((cn[q] * 20 + y) * 20 + xx) * 256 + ccg[q] * 8;
            const unsigned lo =
                lds_base + (unsigned)(buf * BUFH + ccol[q] * BCOL_PITCH + ccg[q] * 8) * 2u;
            asm volatile("global_load_async_to_lds_b128 %0, %1, off"
                         :: "v"(lo), "v"((unsigned long long)(unsigned long long)gp)
                         : "memory");
        }
    };

    const _Float16* ArowA = Wr + ((wave << 4) + lm) * K2;           // co tile w
    const _Float16* ArowB = Wr + ((wave << 4) + 128 + lm) * K2;     // co tile w+8
    v8f accA = {};
    v8f accB = {};

    issue_copy(0, 0);
    for (int p = 0; p < 81; ++p) {
        const int buf = p & 1;
        if (p < 80) {
            issue_copy(p + 1, buf ^ 1);
            WAIT_ASYNC(2);               // copy(p) complete (in-order async)
        } else {
            WAIT_ASYNC(0);
        }
        __syncthreads();                 // slab p visible to all waves

        const _Float16* bbase = &bstage[buf * BUFH + lm * BCOL_PITCH + hi * 8];
#pragma unroll
        for (int cc = 0; cc < 8; ++cc) {
            const int kc   = p * 8 + cc;
            const int aidx = (kc << 5) + (hi << 3);
            v16h afA, afB, bf;
            ((v8h*)&afA)[0] = *(const v8h*)(ArowA + aidx);
            ((v8h*)&afA)[1] = *(const v8h*)(ArowA + aidx + 16);
            ((v8h*)&afB)[0] = *(const v8h*)(ArowB + aidx);
            ((v8h*)&afB)[1] = *(const v8h*)(ArowB + aidx + 16);
            ((v8h*)&bf)[0]  = *(const v8h*)(bbase + cc * 32);
            ((v8h*)&bf)[1]  = *(const v8h*)(bbase + cc * 32 + 16);
            accA = __builtin_amdgcn_wmma_f32_16x16x32_f16(
                false, afA, false, bf, (short)0, accA, false, false);
            accB = __builtin_amdgcn_wmma_f32_16x16x32_f16(
                false, afB, false, bf, (short)0, accB, false, false);
        }
        __syncthreads();                 // all waves done with slab p
    }

    // epilogue: rows co = m0 + hi*8 + r  ->  cap = (m0+hi*8)/8, d = r
#pragma unroll
    for (int half = 0; half < 2; ++half) {
        const int m0      = (wave << 4) + half * 128;
        const int co_base = m0 + (hi << 3);
        const int cap     = co_base >> 3;
        const int i       = cap * 36 + pos;
        v8f bv  = *(const v8f*)(bias + co_base);
        v8f res = (half ? accB : accA) + bv;
        *(v8f*)(u_pre + ((size_t)n * NCAPS + i) * CAPD) = res;
    }
}

// ---------------------------------------------------------------------------
// Kernel 5: squash u over dim=8, write f16 (WMMA A input for u_hat)
// ---------------------------------------------------------------------------
__global__ __launch_bounds__(256)
void k_squash_u(const float* __restrict__ u_pre, _Float16* __restrict__ u16) {
    const int idx = blockIdx.x * 256 + threadIdx.x;    // < 512*1152
    const float* p = u_pre + (size_t)idx * CAPD;
    float lensq = 0.f;
#pragma unroll
    for (int k = 0; k < CAPD; ++k) lensq += p[k] * p[k];
    const float len   = sqrtf(lensq);
    const float scale = lensq / (1.0f + lensq) / len;
#pragma unroll
    for (int k = 0; k < CAPD; ++k)
        u16[(size_t)idx * CAPD + k] = (_Float16)(p[k] * scale);
}

// ---------------------------------------------------------------------------
// Kernel 6: u_hat[n][i][o] = u[n][i][:8] @ caps_w[i][:8][o]
// Batched per-capsule GEMM via WMMA f16 (K padded 8->32 with zeros in regs).
// ---------------------------------------------------------------------------
__global__ __launch_bounds__(256)
void k_uhat_wmma(const _Float16* __restrict__ u16, const _Float16* __restrict__ ct,
                 float* __restrict__ u_hat) {
    const int lane = threadIdx.x & 31;
    const int wave = threadIdx.x >> 5;
    const int tile = blockIdx.x * 8 + wave;            // < 368640
    const int i    = tile / 320;
    const int rem  = tile - i * 320;
    const int mt   = rem / 10;
    const int ot   = rem - mt * 10;
    const int n0   = mt << 4;
    const int o0   = ot << 4;
    const int lm   = lane & 15;
    const int hi   = lane >> 4;

    v16h af = {};
    v16h bf = {};
    if (!hi) {
        ((v8h*)&af)[0] = *(const v8h*)(u16 + ((size_t)(n0 + lm) * NCAPS + i) * CAPD);
        ((v8h*)&bf)[0] = *(const v8h*)(ct + ((size_t)i * 160 + o0 + lm) * CAPD);
    }
    v8f acc = {};
    acc = __builtin_amdgcn_wmma_f32_16x16x32_f16(
        false, af, false, bf, (short)0, acc, false, false);

#pragma unroll
    for (int r = 0; r < 8; ++r) {
        const int nn = n0 + r + (hi << 3);
        u_hat[((size_t)nn * NCAPS + i) * 160 + o0 + lm] = acc[r];
    }
}

// ---------------------------------------------------------------------------
// Kernel 7: dynamic routing (3 iterations) + outputs. One block per image.
// b logits live in registers (each thread owns i = t, t+256, ...).
// ---------------------------------------------------------------------------
__global__ __launch_bounds__(256)
void k_routing(const float* __restrict__ u_hat, const float* __restrict__ b_prior,
               float* __restrict__ out_v, float* __restrict__ out_p) {
    __shared__ float c_lds[NCAPS * OCAPS];             // 46080 B
    __shared__ float v_lds[160];
    __shared__ float s_lds[160];

    const int n = blockIdx.x;
    const int t = threadIdx.x;
    const float* uhn = u_hat + (size_t)n * NCAPS * 160;

    float breg[5][OCAPS];
    for (int q = 0; q < 5; ++q) {
        const int i = t + 256 * q;
        if (i < NCAPS) {
            float m = -1e30f;
#pragma unroll
            for (int jj = 0; jj < OCAPS; ++jj) {
                breg[q][jj] = b_prior[i * OCAPS + jj];
                m = fmaxf(m, breg[q][jj]);
            }
            float sum = 0.f, e[OCAPS];
#pragma unroll
            for (int jj = 0; jj < OCAPS; ++jj) { e[jj] = __expf(breg[q][jj] - m); sum += e[jj]; }
            const float inv = 1.0f / sum;
#pragma unroll
            for (int jj = 0; jj < OCAPS; ++jj) c_lds[i * OCAPS + jj] = e[jj] * inv;
        }
    }
    __syncthreads();

    for (int it = 0; it <= 3; ++it) {
        // s[j][d] = sum_i c[i][j] * u_hat[n][i][j*16+d]
        if (t < 160) {
            const int jj = t >> 4;
            float s = 0.f;
            for (int i = 0; i < NCAPS; ++i)
                s += c_lds[i * OCAPS + jj] * uhn[(size_t)i * 160 + t];
            s_lds[t] = s;
        }
        __syncthreads();
        if (t < 160) {
            const int jj = t >> 4;
            float l = 0.f;
#pragma unroll
            for (int d = 0; d < 16; ++d) { float sv = s_lds[jj * 16 + d]; l += sv * sv; }
            const float len = sqrtf(l);
            const float sc  = l / (1.0f + l) / len;
            v_lds[t] = s_lds[t] * sc;
        }
        __syncthreads();

        if (it == 3) break;

        for (int q = 0; q < 5; ++q) {
            const int i = t + 256 * q;
            if (i < NCAPS) {
                const float* uh = uhn + (size_t)i * 160;
#pragma unroll
                for (int jj = 0; jj < OCAPS; ++jj) {
                    float a = 0.f;
#pragma unroll
                    for (int d = 0; d < 16; ++d) a += uh[jj * 16 + d] * v_lds[jj * 16 + d];
                    breg[q][jj] += a;
                }
                float m = -1e30f;
#pragma unroll
                for (int jj = 0; jj < OCAPS; ++jj) m = fmaxf(m, breg[q][jj]);
                float sum = 0.f, e[OCAPS];
#pragma unroll
                for (int jj = 0; jj < OCAPS; ++jj) { e[jj] = __expf(breg[q][jj] - m); sum += e[jj]; }
                const float inv = 1.0f / sum;
#pragma unroll
                for (int jj = 0; jj < OCAPS; ++jj) c_lds[i * OCAPS + jj] = e[jj] * inv;
            }
        }
        __syncthreads();
    }

    if (t < 160) out_v[n * 160 + t] = v_lds[t];
    if (t < OCAPS) {
        float l = 0.f;
#pragma unroll
        for (int d = 0; d < 16; ++d) { float vv = v_lds[t * 16 + d]; l += vv * vv; }
        out_p[n * OCAPS + t] = sqrtf(l);
    }
}

// ---------------------------------------------------------------------------
extern "C" void kernel_launch(void* const* d_in, const int* in_sizes, int n_in,
                              void* d_out, int out_size, void* d_ws, size_t ws_size,
                              hipStream_t stream) {
    const float* x       = (const float*)d_in[0];
    const float* conv1_w = (const float*)d_in[1];
    const float* conv1_b = (const float*)d_in[2];
    const float* conv2_w = (const float*)d_in[3];
    const float* conv2_b = (const float*)d_in[4];
    const float* caps_w  = (const float*)d_in[5];
    const float* b_prior = (const float*)d_in[6];

    char* ws = (char*)d_ws;
    size_t off = 0;
    auto take = [&](size_t bytes) { char* p = ws + off; off = (off + bytes + 255) & ~(size_t)255; return p; };

    _Float16* h1    = (_Float16*)take((size_t)N_IMG * 20 * 20 * 256 * 2);     // 104.9 MB
    _Float16* Wr    = (_Float16*)take((size_t)256 * K2 * 2);                  // 10.6 MB
    _Float16* cwt   = (_Float16*)take((size_t)NCAPS * 160 * 8 * 2);           // 2.9 MB
    float*    u_pre = (float*)   take((size_t)N_IMG * NCAPS * CAPD * 4);      // 18.9 MB
    _Float16* u16   = (_Float16*)take((size_t)N_IMG * NCAPS * CAPD * 2);      // 9.4 MB
    float*    u_hat = (float*)   take((size_t)N_IMG * NCAPS * 160 * 4);       // 377.5 MB

    float* out_v = (float*)d_out;                  // [512,10,16]
    float* out_p = out_v + N_IMG * 160;            // [512,10]

    k_conv1<<<N_IMG, 256, 0, stream>>>(x, conv1_w, conv1_b, h1);
    k_repack_w2<<<(256 * K2) / 256, 256, 0, stream>>>(conv2_w, Wr);
    k_repack_cw<<<(NCAPS * 160 * 8) / 256, 256, 0, stream>>>(caps_w, cwt);

    // conv2 implicit GEMM: one block per 16-column tile, async-LDS staged B
    k_conv2_wmma<<<COLS2 / 16, 256, 0, stream>>>(Wr, h1, conv2_b, u_pre);

    k_squash_u<<<(N_IMG * NCAPS) / 256, 256, 0, stream>>>(u_pre, u16);
    k_uhat_wmma<<<(NCAPS * 32 * 10) / 8, 256, 0, stream>>>(u16, cwt, u_hat);
    k_routing<<<N_IMG, 256, 0, stream>>>(u_hat, b_prior, out_v, out_p);
}
